// Multi_head_attention_71554155151468
// MI455X (gfx1250) — matrix operand order repaired and checked by reading, not compiled, not run on hardware
//
#include <hip/hip_runtime.h>
#include <hip/hip_bf16.h>
#include <stdint.h>

typedef __attribute__((ext_vector_type(16))) __bf16 v16bf;
typedef __attribute__((ext_vector_type(8)))  float  v8f;

#define NB    4
#define SEQ   2048
#define HID   300
#define HIDP  512
#define NH    8
#define DH    64
#define KP    320   // HID zero-padded to multiple of 32

// ---------------- CDNA5 async global->LDS staging ----------------
// GLOBAL_LOAD_ASYNC_TO_LDS_B128: LDS address from a 32-bit VGPR, global address
// from a VGPR pair; tracked with ASYNCcnt. INST_OFFSET adds to BOTH addresses
// so one address pair serves a
// whole row segment via offset immediates.

#define ASYNC_B128(lds, g, OFF)                                           \
  asm volatile("global_load_async_to_lds_b128 %0, %1, off offset:" #OFF   \
               :                                                          \
               : "v"(lds), "v"(g)                                         \
               : "memory")

static __device__ __forceinline__ void stage32(uint32_t l, const void* g) {
  ASYNC_B128(l, g, 0);
  ASYNC_B128(l, g, 16);
}
static __device__ __forceinline__ void stage64(uint32_t l, const void* g) {
  ASYNC_B128(l, g, 0);
  ASYNC_B128(l, g, 16);
  ASYNC_B128(l, g, 32);
  ASYNC_B128(l, g, 48);
}

static __device__ __forceinline__ void wait_async0() {
  asm volatile("s_wait_asynccnt 0x0" ::: "memory");
}

static __device__ __forceinline__ uint32_t lds_off(const void* p) {
  return (uint32_t)(size_t)p;  // LDS offset = low 32 bits of the address
}

// ---------------- WMMA helpers (CDNA5 bf16 path) ----------------

static __device__ __forceinline__ v8f wmma_bf16(v16bf a, v16bf b, v8f c) {
  // (neg_a, A, neg_b, B, c_mod, C, reuse_a, reuse_b)
  return __builtin_amdgcn_wmma_f32_16x16x32_bf16(false, a, false, b, (short)0, c,
                                                 false, false);
}

// A-matrix 16x32 bf16 fragment from row-major LDS tile (ISA 7.12.2):
// lanes 0-15: M=lane, K = 0..7 | 16..23 ; lanes 16-31: M=lane-16, K = 8..15 | 24..31
static __device__ __forceinline__ v16bf frag_a(const __bf16* p, int row0, int k0,
                                               int ld) {
  const int lane = threadIdx.x & 31;
  const int m    = lane & 15;
  const int kb   = (lane >> 4) << 3;  // 0 or 8
  const __bf16* s = p + (row0 + m) * ld + k0 + kb;
  union { v16bf v; uint4 q[2]; } u;
  u.q[0] = *(const uint4*)(s);
  u.q[1] = *(const uint4*)(s + 16);
  return u.v;
}

// B-matrix 32x16 bf16 fragment from a *transposed* [N][K] LDS tile:
// lanes 0-15: N=lane, K = 0..7 | 16..23 ; lanes 16-31: N=lane-16, K = 8..15 | 24..31
static __device__ __forceinline__ v16bf frag_bT(const __bf16* p, int col0, int k0,
                                                int ld) {
  const int lane = threadIdx.x & 31;
  const int n    = lane & 15;
  const int kb   = (lane >> 4) << 3;  // 0 or 8
  const __bf16* s = p + (col0 + n) * ld + k0 + kb;
  union { v16bf v; uint4 q[2]; } u;
  u.q[0] = *(const uint4*)(s);
  u.q[1] = *(const uint4*)(s + 16);
  return u.v;
}

// ---------------- Prep kernels: one-shot convert/transpose/pad ----------------

// de/en f32 [8192][300] -> bf16 [8192][KP] (zero-padded)
__global__ __launch_bounds__(256) void prep_act_kernel(
    const float* __restrict__ de, const float* __restrict__ en,
    __bf16* __restrict__ deb, __bf16* __restrict__ enb) {
  const int idx = blockIdx.x * 256 + threadIdx.x;  // one float4 group
  const int r  = idx / (KP / 4);
  const int c0 = (idx % (KP / 4)) * 4;
  const float* src = blockIdx.y ? en : de;
  __bf16* dst = blockIdx.y ? enb : deb;
  float4 f = (c0 < HID) ? *(const float4*)(src + (size_t)r * HID + c0)
                        : make_float4(0.f, 0.f, 0.f, 0.f);
  __bf16* d = dst + (size_t)r * KP + c0;
  d[0] = (__bf16)f.x; d[1] = (__bf16)f.y; d[2] = (__bf16)f.z; d[3] = (__bf16)f.w;
}

// WQ/WK/WV f32 [300][512] -> bf16 transposed [512][KP] (zero-padded K)
__global__ __launch_bounds__(256) void prep_w_kernel(
    const float* __restrict__ WQ, const float* __restrict__ WK,
    const float* __restrict__ WV, __bf16* __restrict__ wqt,
    __bf16* __restrict__ wkt, __bf16* __restrict__ wvt) {
  const int idx = blockIdx.x * 256 + threadIdx.x;  // 0 .. 512*KP-1
  const int n = idx / KP;
  const int k = idx % KP;
  const float* W = (blockIdx.y == 0) ? WQ : (blockIdx.y == 1) ? WK : WV;
  __bf16* D = (blockIdx.y == 0) ? wqt : (blockIdx.y == 1) ? wkt : wvt;
  float v = (k < HID) ? W[(size_t)k * HIDP + n] : 0.f;
  D[(size_t)n * KP + k] = (__bf16)v;
}

// WO f32 [512][300] -> bf16 transposed [KP rows n][512 k] (rows >=300 zero)
__global__ __launch_bounds__(256) void prep_wo_kernel(const float* __restrict__ WO,
                                                      __bf16* __restrict__ wot) {
  const int idx = blockIdx.x * 256 + threadIdx.x;  // 0 .. KP*512-1
  const int n = idx / HIDP;
  const int k = idx % HIDP;
  float v = (n < HID) ? WO[(size_t)k * HID + n] : 0.f;
  wot[(size_t)n * HIDP + k] = (__bf16)v;
}

// ---------------- Kernel 1: fused QKV projection GEMM ----------------
// mode 0: Q = de@WQ -> q_ws[bh][sd][64]
// mode 1: K = en@WK -> k_ws[bh][se][64]   (natural; attn reads it as B^T tile)
// mode 2: V = en@WV -> v_ws[bh][64][se]   (transposed + K-permuted per 64-block)
// head de-interleave: feature n -> h = n%8, c = n/8 (matches reshape(...,-1,8))

#define LDA1  40  // 32-wide A tile, +8 pad (multiple of 8)
#define LDB1T 40  // transposed B tile [64 n][32 k], +8 pad
#define NKCH  (KP / 32)   // 10 (even)

__global__ __launch_bounds__(128) void qkv_proj_kernel(
    const __bf16* __restrict__ enb, const __bf16* __restrict__ deb,
    const __bf16* __restrict__ wqt, const __bf16* __restrict__ wkt,
    const __bf16* __restrict__ wvt, __bf16* __restrict__ q_ws,
    __bf16* __restrict__ k_ws, __bf16* __restrict__ v_ws) {
  __shared__ __bf16 ldsA[2][64 * LDA1];
  __shared__ __bf16 ldsBT[2][64 * LDB1T];

  const int tid  = threadIdx.x;
  const int wave = tid >> 5;
  const int lane = tid & 31;
  const int m0   = blockIdx.x * 64;
  const int n0   = blockIdx.y * 64;
  const int mode = blockIdx.z;

  const __bf16* src = (mode == 0) ? deb : enb;
  const __bf16* Wt  = (mode == 0) ? wqt : (mode == 1) ? wkt : wvt;

  const int r8  = tid >> 1;
  const int c16 = (tid & 1) * 16;
  const __bf16* ga0 = src + (size_t)(m0 + r8) * KP + c16;
  const __bf16* gb0 = Wt + (size_t)(n0 + r8) * KP + c16;
  const uint32_t laA0 = lds_off(ldsA[0] + r8 * LDA1 + c16);
  const uint32_t laA1 = lds_off(ldsA[1] + r8 * LDA1 + c16);
  const uint32_t laB0 = lds_off(ldsBT[0] + r8 * LDB1T + c16);
  const uint32_t laB1 = lds_off(ldsBT[1] + r8 * LDB1T + c16);

  v8f acc[4];
#pragma unroll
  for (int t = 0; t < 4; ++t)
#pragma unroll
    for (int i = 0; i < 8; ++i) acc[t][i] = 0.0f;

  auto compute = [&](const __bf16* pa, const __bf16* pb) {
    v16bf a = frag_a(pa, wave * 16, 0, LDA1);
#pragma unroll
    for (int t = 0; t < 4; ++t)
      acc[t] = wmma_bf16(a, frag_bT(pb, t * 16, 0, LDB1T), acc[t]);
  };

  // explicit ping-pong double buffering (NKCH is even)
  stage32(laA0, ga0);
  stage32(laB0, gb0);
  wait_async0();
  __syncthreads();
  for (int i = 0; i < NKCH; i += 2) {
    stage32(laA1, ga0 + (i + 1) * 32);
    stage32(laB1, gb0 + (i + 1) * 32);
    compute(ldsA[0], ldsBT[0]);
    wait_async0();
    __syncthreads();
    if (i + 2 < NKCH) {
      stage32(laA0, ga0 + (i + 2) * 32);
      stage32(laB0, gb0 + (i + 2) * 32);
    }
    compute(ldsA[1], ldsBT[1]);
    wait_async0();
    __syncthreads();
  }

  const int hi = lane >> 4, ln = lane & 15;
#pragma unroll
  for (int t = 0; t < 4; ++t) {
#pragma unroll
    for (int r = 0; r < 8; ++r) {
      int m    = r + 8 * hi;
      int grow = m0 + wave * 16 + m;   // 0..8191
      int gcol = n0 + t * 16 + ln;     // 0..511
      int b = grow >> 11, s = grow & (SEQ - 1);
      int h = gcol & 7, c = gcol >> 3;
      int bh = b * NH + h;
      __bf16 val = (__bf16)acc[t][r];
      if (mode == 0) {
        q_ws[((size_t)(bh * SEQ + s)) * DH + c] = val;
      } else if (mode == 1) {
        k_ws[((size_t)(bh * SEQ + s)) * DH + c] = val;
      } else {
        // permute se within each 64-block: pi(j) = (j%16)*4 + j/16, matching
        // the contiguous-column P layout used by the attention kernel's P@V.
        int sj = s & 63;
        int sp = (s & ~63) + ((sj & 15) << 2) + (sj >> 4);
        v_ws[((size_t)(bh * DH + c)) * SEQ + sp] = val;
      }
    }
  }
}

// ---------------- Kernel 2: flash attention per (64 Q-rows, b*h) ----------------

#define LDT 72  // 64-wide tiles, +8 pad (multiple of 8)

__global__ __launch_bounds__(128) void attn_kernel(
    const __bf16* __restrict__ q_ws, const __bf16* __restrict__ k_ws,
    const __bf16* __restrict__ v_ws, const uint8_t* __restrict__ mask,
    __bf16* __restrict__ a_ws) {
  __shared__ __bf16 ldsQ[64 * LDT];      // Q tile; reused as per-wave P scratch
  __shared__ __bf16 ldsKT[2][64 * LDT];  // [n=se_local][k=c]  (= natural K rows)
  __shared__ __bf16 ldsVT[2][64 * LDT];  // [n=c][k=pi(se_local)] (= rows of v_ws)
  __shared__ uint8_t ldsM[2][64 * 64];   // mask tiles

  const int tid  = threadIdx.x;
  const int wave = tid >> 5;
  const int lane = tid & 31;
  const int hi = lane >> 4, ln = lane & 15;

  const int bh  = blockIdx.y;        // b*8 + h
  const int sd0 = blockIdx.x * 64;

  const __bf16* qp = q_ws + (size_t)bh * SEQ * DH;
  const __bf16* kp = k_ws + (size_t)bh * SEQ * DH;
  const __bf16* vp = v_ws + (size_t)bh * DH * SEQ;

  const int r64 = tid >> 1;
  const int h32 = (tid & 1) * 32;  // element offset 0/32 within a 64-row

  // per-thread staging bases (advance by tile in the loop)
  const __bf16* gkr = kp + (size_t)r64 * DH + h32;       // + it*64*DH walks rows
  const __bf16* gvr = vp + (size_t)r64 * SEQ + h32;      // + it*64 walks cols
  const uint8_t* gmr = mask + (size_t)(sd0 + r64) * SEQ + h32;
  const uint32_t lk0 = lds_off(ldsKT[0] + r64 * LDT + h32);
  const uint32_t lk1 = lds_off(ldsKT[1] + r64 * LDT + h32);
  const uint32_t lv0 = lds_off(ldsVT[0] + r64 * LDT + h32);
  const uint32_t lv1 = lds_off(ldsVT[1] + r64 * LDT + h32);
  const uint32_t lm0 = lds_off(ldsM[0] + r64 * 64 + h32);
  const uint32_t lm1 = lds_off(ldsM[1] + r64 * 64 + h32);

  // stage Q tile together with first K/V/mask tile
  stage64(lds_off(ldsQ + r64 * LDT + h32), qp + (size_t)(sd0 + r64) * DH + h32);
  stage64(lk0, gkr);
  stage64(lv0, gvr);
  stage32(lm0, gmr);
  wait_async0();
  __syncthreads();
  v16bf qfrag[2];
  qfrag[0] = frag_a(ldsQ, wave * 16, 0, LDT);
  qfrag[1] = frag_a(ldsQ, wave * 16, 32, LDT);
  __syncthreads();  // Q fragments in registers; ldsQ becomes P scratch

  const float scale = 0.05773502691896258f;  // 1/sqrt(300): ref scales by hidden dim
  float mrow[8], lrow[8];
  v8f oacc[4];
#pragma unroll
  for (int r = 0; r < 8; ++r) { mrow[r] = -3.0e38f; lrow[r] = 0.0f; }
#pragma unroll
  for (int t = 0; t < 4; ++t)
#pragma unroll
    for (int i = 0; i < 8; ++i) oacc[t][i] = 0.0f;

  __bf16* pbuf = ldsQ + wave * 16 * LDT;

  auto compute = [&](const __bf16* pk, const __bf16* pv, const uint8_t* pm) {
    // S = (Q K^T) for this wave's 16 rows x 64 keys
    v8f s[4];
#pragma unroll
    for (int t = 0; t < 4; ++t)
#pragma unroll
      for (int i = 0; i < 8; ++i) s[t][i] = 0.0f;
#pragma unroll
    for (int kc = 0; kc < 2; ++kc)
#pragma unroll
      for (int t = 0; t < 4; ++t)
        s[t] = wmma_bf16(qfrag[kc], frag_bT(pk, t * 16, kc * 32, LDT), s[t]);

    // scale + mask (mask true -> -1e9, matching jnp.where(mask, -1e9, e))
#pragma unroll
    for (int t = 0; t < 4; ++t) {
#pragma unroll
      for (int r = 0; r < 8; ++r) {
        int m = wave * 16 + r + 8 * hi;
        int n = t * 16 + ln;
        float v = s[t][r] * scale;
        if (pm[m * 64 + n]) v = -1.0e9f;
        s[t][r] = v;
      }
    }

    // online softmax: per-row (vgpr index) stats, reduce over 16-lane halves
#pragma unroll
    for (int r = 0; r < 8; ++r) {
      float mx = fmaxf(fmaxf(s[0][r], s[1][r]), fmaxf(s[2][r], s[3][r]));
#pragma unroll
      for (int off = 1; off < 16; off <<= 1)
        mx = fmaxf(mx, __shfl_xor(mx, off, 32));
      float mnew = fmaxf(mrow[r], mx);
      float corr = __expf(mrow[r] - mnew);
      float sum = 0.0f;
#pragma unroll
      for (int t = 0; t < 4; ++t) {
        float p = __expf(s[t][r] - mnew);
        s[t][r] = p;
        sum += p;
      }
#pragma unroll
      for (int off = 1; off < 16; off <<= 1) sum += __shfl_xor(sum, off, 32);
      lrow[r] = lrow[r] * corr + sum;
      mrow[r] = mnew;
#pragma unroll
      for (int t = 0; t < 4; ++t) oacc[t][r] *= corr;
    }

    // P -> per-wave LDS scratch in pi-permuted column order: lane's 4 t-values
    // are contiguous (pi(t*16+ln) = ln*4+t) -> single b64 store per row.
#pragma unroll
    for (int r = 0; r < 8; ++r) {
      __bf16* d = pbuf + (r + 8 * hi) * LDT + ln * 4;
      d[0] = (__bf16)s[0][r];
      d[1] = (__bf16)s[1][r];
      d[2] = (__bf16)s[2][r];
      d[3] = (__bf16)s[3][r];
    }

#pragma unroll
    for (int kc = 0; kc < 2; ++kc) {
      v16bf ap = frag_a(pbuf, 0, kc * 32, LDT);
#pragma unroll
      for (int t = 0; t < 4; ++t)
        oacc[t] = wmma_bf16(ap, frag_bT(pv, t * 16, kc * 32, LDT), oacc[t]);
    }
  };

  // explicit ping-pong over 32 key tiles (even count)
  for (int it = 0; it < SEQ / 64; it += 2) {
    stage64(lk1, gkr + (size_t)(it + 1) * 64 * DH);
    stage64(lv1, gvr + (it + 1) * 64);
    stage32(lm1, gmr + (it + 1) * 64);
    compute(ldsKT[0], ldsVT[0], ldsM[0]);
    wait_async0();
    __syncthreads();
    if (it + 2 < SEQ / 64) {
      stage64(lk0, gkr + (size_t)(it + 2) * 64 * DH);
      stage64(lv0, gvr + (it + 2) * 64);
      stage32(lm0, gmr + (it + 2) * 64);
    }
    compute(ldsKT[1], ldsVT[1], ldsM[1]);
    wait_async0();
    __syncthreads();
  }

  // normalize and store in head-concat layout a[b][sd][h*64 + c]
  const int b = bh >> 3, h = bh & 7;
#pragma unroll
  for (int t = 0; t < 4; ++t) {
#pragma unroll
    for (int r = 0; r < 8; ++r) {
      int m  = r + 8 * hi;
      int sd = sd0 + wave * 16 + m;
      int c  = t * 16 + ln;
      float val = oacc[t][r] / lrow[r];
      a_ws[((size_t)(b * SEQ + sd)) * HIDP + h * DH + c] = (__bf16)val;
    }
  }
}

// ---------------- Kernel 3: output projection a @ WO (N=300 guarded) ----------------

__global__ __launch_bounds__(128) void out_proj_kernel(
    const __bf16* __restrict__ a_ws, const __bf16* __restrict__ wot,
    float* __restrict__ out) {
  __shared__ __bf16 ldsA[2][64 * LDA1];
  __shared__ __bf16 ldsBT[2][64 * LDB1T];

  const int tid  = threadIdx.x;
  const int wave = tid >> 5;
  const int lane = tid & 31;
  const int m0 = blockIdx.x * 64;
  const int n0 = blockIdx.y * 64;

  const int r8  = tid >> 1;
  const int c16 = (tid & 1) * 16;
  const __bf16* ga0 = a_ws + (size_t)(m0 + r8) * HIDP + c16;
  const __bf16* gb0 = wot + (size_t)(n0 + r8) * HIDP + c16;
  const uint32_t laA0 = lds_off(ldsA[0] + r8 * LDA1 + c16);
  const uint32_t laA1 = lds_off(ldsA[1] + r8 * LDA1 + c16);
  const uint32_t laB0 = lds_off(ldsBT[0] + r8 * LDB1T + c16);
  const uint32_t laB1 = lds_off(ldsBT[1] + r8 * LDB1T + c16);

  v8f acc[4];
#pragma unroll
  for (int t = 0; t < 4; ++t)
#pragma unroll
    for (int i = 0; i < 8; ++i) acc[t][i] = 0.0f;

  auto compute = [&](const __bf16* pa, const __bf16* pb) {
    v16bf a = frag_a(pa, wave * 16, 0, LDA1);
#pragma unroll
    for (int t = 0; t < 4; ++t)
      acc[t] = wmma_bf16(a, frag_bT(pb, t * 16, 0, LDB1T), acc[t]);
  };

  const int NK = HIDP / 32;  // 16 (even)
  stage32(laA0, ga0);
  stage32(laB0, gb0);
  wait_async0();
  __syncthreads();
  for (int i = 0; i < NK; i += 2) {
    stage32(laA1, ga0 + (i + 1) * 32);
    stage32(laB1, gb0 + (i + 1) * 32);
    compute(ldsA[0], ldsBT[0]);
    wait_async0();
    __syncthreads();
    if (i + 2 < NK) {
      stage32(laA0, ga0 + (i + 2) * 32);
      stage32(laB0, gb0 + (i + 2) * 32);
    }
    compute(ldsA[1], ldsBT[1]);
    wait_async0();
    __syncthreads();
  }

  const int hi = lane >> 4, ln = lane & 15;
#pragma unroll
  for (int t = 0; t < 4; ++t) {
#pragma unroll
    for (int r = 0; r < 8; ++r) {
      int grow = m0 + wave * 16 + r + 8 * hi;
      int gcol = n0 + t * 16 + ln;
      if (gcol < HID) out[(size_t)grow * HID + gcol] = acc[t][r];
    }
  }
}

// ---------------- Launch ----------------

extern "C" void kernel_launch(void* const* d_in, const int* in_sizes, int n_in,
                              void* d_out, int out_size, void* d_ws, size_t ws_size,
                              hipStream_t stream) {
  (void)in_sizes; (void)n_in; (void)out_size; (void)ws_size;

  const float*   en   = (const float*)d_in[0];
  const float*   de   = (const float*)d_in[1];
  const uint8_t* mask = (const uint8_t*)d_in[2];  // jnp.bool_ -> 1 byte/elem
  const float*   WQ   = (const float*)d_in[3];
  const float*   WK   = (const float*)d_in[4];
  const float*   WV   = (const float*)d_in[5];
  const float*   WO   = (const float*)d_in[6];
  float*         out  = (float*)d_out;

  // workspace layout (bf16):
  //  q|k|v|a: 8 MiB each; deb/enb: 8192*KP; wq/wk/wv_t: 512*KP; wot: KP*512
  char* ws = (char*)d_ws;
  const size_t seg  = (size_t)NB * NH * SEQ * DH * sizeof(__bf16);  // 8 MiB
  const size_t aseg = (size_t)NB * SEQ * KP * sizeof(__bf16);       // 5 MiB
  const size_t wseg = (size_t)HIDP * KP * sizeof(__bf16);           // 320 KiB
  __bf16* q_ws = (__bf16*)(ws);
  __bf16* k_ws = (__bf16*)(ws + seg);
  __bf16* v_ws = (__bf16*)(ws + 2 * seg);
  __bf16* a_ws = (__bf16*)(ws + 3 * seg);
  __bf16* deb  = (__bf16*)(ws + 4 * seg);
  __bf16* enb  = (__bf16*)(ws + 4 * seg + aseg);
  __bf16* wqt  = (__bf16*)(ws + 4 * seg + 2 * aseg);
  __bf16* wkt  = (__bf16*)(ws + 4 * seg + 2 * aseg + wseg);
  __bf16* wvt  = (__bf16*)(ws + 4 * seg + 2 * aseg + 2 * wseg);
  __bf16* wot  = (__bf16*)(ws + 4 * seg + 2 * aseg + 3 * wseg);

  prep_act_kernel<<<dim3((NB * SEQ * (KP / 4)) / 256, 2), 256, 0, stream>>>(
      de, en, deb, enb);
  prep_w_kernel<<<dim3((HIDP * KP) / 256, 3), 256, 0, stream>>>(WQ, WK, WV, wqt,
                                                                wkt, wvt);
  prep_wo_kernel<<<dim3((KP * HIDP) / 256), 256, 0, stream>>>(WO, wot);

  qkv_proj_kernel<<<dim3((NB * SEQ) / 64, HIDP / 64, 3), 128, 0, stream>>>(
      enb, deb, wqt, wkt, wvt, q_ws, k_ws, v_ws);
  attn_kernel<<<dim3(SEQ / 64, NB * NH), 128, 0, stream>>>(q_ws, k_ws, v_ws, mask,
                                                           a_ws);
  out_proj_kernel<<<dim3((NB * SEQ) / 64, 5), 128, 0, stream>>>(a_ws, wot, out);
}
